// BilinearDecoder_386547057343
// MI455X (gfx1250) — compile-verified, matrix operand order checked by reading, not run on hardware
//
#include <hip/hip_runtime.h>
#include <hip/hip_bf16.h>

typedef __attribute__((ext_vector_type(16))) _Float16 v16h;
typedef __attribute__((ext_vector_type(8)))  _Float16 v8h;
typedef __attribute__((ext_vector_type(8)))  float    v8f;

namespace {
constexpr int       Bn = 32;
constexpr int       Ln = 1024;
constexpr int       Hn = 128;
constexpr float     NEG_BIG = 1e8f;
constexpr long long Sz = (long long)Ln * Bn * Ln;   // 33,554,432 elements per output
}

union CFrag { v8f v; float f[8]; };

// ---------------------------------------------------------------------------
// helpers
// ---------------------------------------------------------------------------
// Output index space is 2^25 elements, so a 32-bit low-bias hash suffices:
// 2 mults + 2 xor-shifts instead of splitmix64's three 64-bit multiplies.
__device__ __forceinline__ float rng_uniform32(unsigned x) {
  x ^= 42u * 0x9E3779B9u;               // key = 42
  x *= 0x7FEB352Du; x ^= x >> 15;
  x *= 0x846CA68Bu; x ^= x >> 16;
  return (float)(x >> 8) * (1.0f / 16777216.0f);   // 24-bit uniform in [0,1)
}

// A/B fragment loader: 16 halves per lane = two contiguous 16-byte chunks.
// p must be 16-byte aligned (row stride 256B, k offsets multiples of 16B).
__device__ __forceinline__ v16h load_frag(const _Float16* __restrict__ p) {
  union { v16h v; v8h h8[2]; } u;
  u.h8[0] = *(const v8h*)(p);        // K = base .. base+7   -> global_load_b128
  u.h8[1] = *(const v8h*)(p + 16);   // K = base+16 .. +23   -> global_load_b128
  return u.v;
}

// ---------------------------------------------------------------------------
// Stage 0a: fp32 -> fp16 conversion (enc)
// ---------------------------------------------------------------------------
__global__ void f32_to_f16_kernel(const float* __restrict__ src,
                                  _Float16* __restrict__ dst, int n) {
  int i = blockIdx.x * blockDim.x + threadIdx.x;
  if (i < n) dst[i] = (_Float16)src[i];
}

// ---------------------------------------------------------------------------
// Stage 0b: W fp32 -> Wt fp16 transposed, so stage-1 B-matrix columns
// (= rows of Wt) are contiguous and can use the same b128 fragment loader.
// ---------------------------------------------------------------------------
__global__ void transpose_w_kernel(const float* __restrict__ W,
                                   _Float16* __restrict__ Wt) {
  int i = blockIdx.x * blockDim.x + threadIdx.x;   // over H*H
  if (i < Hn * Hn) {
    int n = i / Hn, h = i % Hn;                    // Wt[n][h] = W[h][n]
    Wt[i] = (_Float16)W[h * Hn + n];
  }
}

// ---------------------------------------------------------------------------
// Stage 1: Y[b,l,n] = sum_h enc[b,l,h] * W[h,n]   (f16 in, f32 acc, f16 out)
// One wave computes one 16x16 tile of the (B*L) x H result.
// ---------------------------------------------------------------------------
__global__ void gemm_xw_kernel(const _Float16* __restrict__ X,   // [B*L, H]
                               const _Float16* __restrict__ Wt,  // [H, H] transposed
                               _Float16* __restrict__ Y)         // [B*L, H]
{
  const int lane = threadIdx.x & 31;
  const int wave = threadIdx.x >> 5;
  const int tile = blockIdx.x * 8 + wave;
  const int ntn  = Hn / 16;                 // 8 column tiles
  const int m0   = (tile / ntn) * 16;
  const int n0   = (tile % ntn) * 16;

  const int ksel = (lane >> 4) << 3;        // 0 for lanes 0-15, 8 for 16-31
  const _Float16* pa = X  + (long long)(m0 + (lane & 15)) * Hn + ksel;
  const _Float16* pb = Wt + (long long)(n0 + (lane & 15)) * Hn + ksel;

  v8f c = {};
#pragma unroll
  for (int k0 = 0; k0 < Hn; k0 += 32) {
    v16h a = load_frag(pa + k0);
    v16h b = load_frag(pb + k0);
    c = __builtin_amdgcn_wmma_f32_16x16x32_f16(false, a, false, b,
                                               (short)0, c, false, false);
  }

  CFrag cf; cf.v = c;
  const int mbase = m0 + ((lane >> 4) << 3);
  const int nout  = n0 + (lane & 15);
#pragma unroll
  for (int r = 0; r < 8; ++r)
    Y[(long long)(mbase + r) * Hn + nout] = (_Float16)cf.f[r];
}

// ---------------------------------------------------------------------------
// Stage 2: logits[b] = Y[b] @ enc[b]^T, fused mask/sigmoid/sample/entropy.
// One wave computes one 16x16 tile of logits[b]; B-fragment columns are rows
// of enc[b] so both operands are contiguous b128 runs (all L2-resident).
//
// Epilogue math (per element, from a single e = exp(-|x|)):
//   t  = rcp(1+e);  p = x>=0 ? t : e*t          (= sigmoid(x))
//   sp = max(x,0) + log(1+e)                    (= softplus(x))
//   entropy = p*sp(-x) + (1-p)*sp(x) = sp - p*x (identity)
// Outputs in transposed [L, B, L] layout: samples | masked | entropy.
// ---------------------------------------------------------------------------
__global__ void bilinear_out_kernel(const _Float16* __restrict__ Y,  // [B, L, H]
                                    const _Float16* __restrict__ X,  // [B, L, H]
                                    const float* __restrict__ lbias,
                                    float* __restrict__ out)
{
  const int lane = threadIdx.x & 31;
  const int wave = threadIdx.x >> 5;
  const long long tile = (long long)blockIdx.x * 8 + wave;

  const int nt = Ln / 16;                        // 64
  const int b  = (int)(tile / (nt * nt));
  const int r2 = (int)(tile % (nt * nt));
  const int m0 = (r2 / nt) * 16;
  const int n0 = (r2 % nt) * 16;

  const int ksel = (lane >> 4) << 3;
  const _Float16* pa =
      Y + (long long)b * Ln * Hn + (long long)(m0 + (lane & 15)) * Hn + ksel;
  const _Float16* pb =
      X + (long long)b * Ln * Hn + (long long)(n0 + (lane & 15)) * Hn + ksel;

  v8f c = {};
#pragma unroll
  for (int k0 = 0; k0 < Hn; k0 += 32) {
    v16h a  = load_frag(pa + k0);
    v16h bf = load_frag(pb + k0);
    c = __builtin_amdgcn_wmma_f32_16x16x32_f16(false, a, false, bf,
                                               (short)0, c, false, false);
  }

  const float bias = lbias[0];
  CFrag cf; cf.v = c;
  const int mbase = m0 + ((lane >> 4) << 3);
  const int n     = n0 + (lane & 15);
#pragma unroll
  for (int r = 0; r < 8; ++r) {
    const int m = mbase + r;                     // logits row l
    float x = cf.f[r] + bias;
    if (m == n) x -= NEG_BIG;                    // diagonal mask

    const float e  = __expf(-fabsf(x));          // one v_exp_f32
    const float t  = __builtin_amdgcn_rcpf(1.0f + e);
    const float p  = (x >= 0.0f) ? t : e * t;    // sigmoid(x)
    const float sp = fmaxf(x, 0.0f) + __logf(1.0f + e);  // softplus(x)
    const float ent = sp - p * x;                // stable Bernoulli entropy

    const long long idx =
        (long long)m * (Bn * Ln) + (long long)b * Ln + n;   // [L, B, L]
    const float smp = (rng_uniform32((unsigned)idx) < p) ? 1.0f : 0.0f;

    // write-once outputs: bypass L2 retention with non-temporal stores
    __builtin_nontemporal_store(smp, out + idx);
    __builtin_nontemporal_store(x,   out + Sz + idx);
    __builtin_nontemporal_store(ent, out + 2 * Sz + idx);
  }
}

// ---------------------------------------------------------------------------
extern "C" void kernel_launch(void* const* d_in, const int* in_sizes, int n_in,
                              void* d_out, int out_size, void* d_ws, size_t ws_size,
                              hipStream_t stream) {
  const float* enc = (const float*)d_in[0];   // [B, L, H] f32
  const float* Wf  = (const float*)d_in[1];   // [H, H]    f32
  const float* lb  = (const float*)d_in[2];   // [1]       f32
  float* out = (float*)d_out;

  // workspace: enc-f16 (8.4MB) | Wt-f16 (32KB) | Y-f16 (8.4MB)
  _Float16* Xh = (_Float16*)d_ws;
  _Float16* Wt = Xh + (size_t)Bn * Ln * Hn;
  _Float16* Yh = Wt + (size_t)Hn * Hn;

  const int nX = Bn * Ln * Hn;
  f32_to_f16_kernel<<<(nX + 255) / 256, 256, 0, stream>>>(enc, Xh, nX);
  const int nW = Hn * Hn;
  transpose_w_kernel<<<(nW + 255) / 256, 256, 0, stream>>>(Wf, Wt);

  // stage 1: (B*L/16) * (H/16) = 16384 wave-tiles, 8 waves / 256-thread block
  gemm_xw_kernel<<<(Bn * Ln / 16) * (Hn / 16) / 8, 256, 0, stream>>>(Xh, Wt, Yh);

  // stage 2: B * (L/16)^2 = 131072 wave-tiles -> 16384 blocks
  bilinear_out_kernel<<<Bn * (Ln / 16) * (Ln / 16) / 8, 256, 0, stream>>>(
      Yh, Xh, lb, out);
}
